// GruSeqModel_85383949844483
// MI455X (gfx1250) — compile-verified, matrix operand order checked by reading, not run on hardware
//
#include <hip/hip_runtime.h>

// GRU on MI455X (gfx1250): bf16 WMMA (v_wmma_f32_16x16x32_bf16), f32 state.
// Each 256-thread block owns 32 batch rows for all 50 steps.
// Round 3: each wave owns a unique 64-col hidden chunk and BOTH 16-row
// M-tiles -> every weight tile is loaded once per WG per step and feeds
// two WMMAs (halves the L2 weight stream vs round 2).

#define IN_DIM 256
#define HID    512
#define SEQ    50
#define BATCH  4096
#define TILE_B 32

typedef __attribute__((ext_vector_type(16))) __bf16 v16bf;
typedef __attribute__((ext_vector_type(8)))  float  v8f;

__device__ __forceinline__ unsigned short f2bf(float f) {
  union { float f; unsigned u; } c; c.f = f;
  unsigned u = c.u;
  u += 0x7FFFu + ((u >> 16) & 1u);          // round-to-nearest-even
  return (unsigned short)(u >> 16);
}

__device__ __forceinline__ v8f splat8(float s) {
  v8f v = { s, s, s, s, s, s, s, s };
  return v;
}

__device__ __forceinline__ v8f wmma_bf16(v16bf a, v16bf b, v8f c) {
  return __builtin_amdgcn_wmma_f32_16x16x32_bf16(false, a, false, b, (short)0, c,
                                                 false, false);
}

__device__ __forceinline__ v16bf ld16(const unsigned short* p) {
  return *(const v16bf*)p;
}

// ---------------------------------------------------------------------------
// Weight packing: W (G x K, row-major f32, used as out = in @ W^T) -> bf16
// tiles in the WMMA B-operand per-lane layout:
//   tile(nt,kt) holds W[nt*16+nn][kt*32+kk];  lane = nn + 16*((kk>>3)&1),
//   within-lane index j = (kk&7) | ((kk>>4)<<3); stored as packed bf16 pairs.
// ---------------------------------------------------------------------------
__global__ void pack_w(const float* __restrict__ W, unsigned* __restrict__ outp,
                       int G, int K) {
  int idx = blockIdx.x * blockDim.x + threadIdx.x;
  int pairsPerRow = K >> 1;
  if (idx >= G * pairsPerRow) return;
  int n = idx / pairsPerRow;
  int k = (idx - n * pairsPerRow) << 1;
  int nt = n >> 4, nn = n & 15;
  int kt = k >> 5, kk = k & 31;
  int lA = nn + (((kk >> 3) & 1) << 4);
  int j  = (kk & 7) | ((kk >> 4) << 3);      // even (k even)
  int numKT = K >> 5;
  unsigned lo = f2bf(W[(long)n * K + k]);
  unsigned hi = f2bf(W[(long)n * K + k + 1]);
  outp[((nt * numKT + kt) << 8) + (lA << 3) + (j >> 1)] = lo | (hi << 16);
}

// Stage x[:, t, :] for this block's 32 rows into LDS as bf16 A-layout tiles.
__device__ __forceinline__ void stage_x(const float* __restrict__ x,
                                        unsigned short* __restrict__ xA_s,
                                        int bBase, int t, int tid) {
#pragma unroll 4
  for (int it = 0; it < 16; ++it) {
    int lin = tid + (it << 8);               // 0..4095
    int m   = lin >> 7;                      // 0..31
    int k   = (lin & 127) << 1;              // even k
    const float2 v = *(const float2*)(x + ((long)(bBase + m) * SEQ + t) * IN_DIM + k);
    int kt = k >> 5, kk = k & 31;
    int lA = (m & 15) + (((kk >> 3) & 1) << 4);
    int j  = (kk & 7) | ((kk >> 4) << 3);
    unsigned pr = (unsigned)f2bf(v.x) | ((unsigned)f2bf(v.y) << 16);
    *(unsigned*)(xA_s + ((((m >> 4) << 3) + kt) << 9) + (lA << 4) + j) = pr;
  }
}

__launch_bounds__(256)
__global__ void gru_fused(const float* __restrict__ x,
                          const float* __restrict__ b_ih,
                          const float* __restrict__ b_hh,
                          const float* __restrict__ b_out,
                          const unsigned short* __restrict__ wihP,
                          const unsigned short* __restrict__ whhP,
                          const unsigned short* __restrict__ woutP,
                          float* __restrict__ out) {
  __shared__ unsigned short xA_s[2 * 8 * 512];    // 16 KB: x_t, 32x256 bf16 A-layout
  __shared__ unsigned short hA_s[2 * 16 * 512];   // 32 KB: h,   32x512 bf16 A-layout

  const int tid   = threadIdx.x;
  const int lane  = tid & 31;
  const int wv    = tid >> 5;        // 8 waves
  const int nb    = wv << 6;         // wave's unique 64-col hidden chunk
  const int bBase = blockIdx.x * TILE_B;
  const int lo16  = lane & 15;
  const int laneB = lane << 4;       // per-lane slice within a tile (bf16 idx)

  // Per-lane gate biases (C-layout: lane%16 selects N within the 16-col tile).
  float biasR[4], biasZ[4], biasNi[4], biasNh[4];
#pragma unroll
  for (int ns = 0; ns < 4; ++ns) {
    int n = nb + (ns << 4) + lo16;
    biasR[ns]  = b_ih[n] + b_hh[n];
    biasZ[ns]  = b_ih[HID + n] + b_hh[HID + n];
    biasNi[ns] = b_ih[2 * HID + n];           // r scales only the hidden part
    biasNh[ns] = b_hh[2 * HID + n];
  }

  // f32 hidden state for this wave's (32 rows x 64 cols) region: [mt][ns].
  v8f h_reg[2][4];
#pragma unroll
  for (int m2 = 0; m2 < 2; ++m2)
#pragma unroll
    for (int ns = 0; ns < 4; ++ns) h_reg[m2][ns] = splat8(0.0f);

  for (int i = tid; i < 2 * 16 * 512; i += 256) hA_s[i] = 0;  // h0 = 0
  stage_x(x, xA_s, bBase, 0, tid);
  __syncthreads();

  const unsigned short* xT0 = xA_s + laneB;              // A tiles, M-tile 0
  const unsigned short* xT1 = xA_s + (8 << 9) + laneB;   // A tiles, M-tile 1
  const unsigned short* hT0 = hA_s + laneB;
  const unsigned short* hT1 = hA_s + (16 << 9) + laneB;

#pragma unroll 1
  for (int t = 0; t < SEQ; ++t) {
    // ---- GEMM + gates: read xA_s (x_t) and hA_s (h_{t-1}) ----
#pragma unroll
    for (int ns = 0; ns < 4; ++ns) {
      const int nt = (nb >> 4) + ns;          // hidden-col tile index (0..31)
      v8f aR0 = splat8(biasR[ns]),  aR1 = splat8(biasR[ns]);
      v8f aZ0 = splat8(biasZ[ns]),  aZ1 = splat8(biasZ[ns]);
      v8f aI0 = splat8(biasNi[ns]), aI1 = splat8(biasNi[ns]);
      v8f aH0 = splat8(biasNh[ns]), aH1 = splat8(biasNh[ns]);

      { // gi part: K = 256 over x_t (8 K-tiles); each B tile feeds 2 WMMAs
        const unsigned short* pR = wihP + (((nt)      * 8) << 9) + laneB;
        const unsigned short* pZ = wihP + (((nt + 32) * 8) << 9) + laneB;
        const unsigned short* pN = wihP + (((nt + 64) * 8) << 9) + laneB;
#pragma unroll 1
        for (int kt = 0; kt < 8; ++kt) {
          const int o = kt << 9;
          v16bf a0 = ld16(xT0 + o);
          v16bf a1 = ld16(xT1 + o);
          v16bf bR = ld16(pR + o), bZ = ld16(pZ + o), bN = ld16(pN + o);
          aR0 = wmma_bf16(a0, bR, aR0);  aR1 = wmma_bf16(a1, bR, aR1);
          aZ0 = wmma_bf16(a0, bZ, aZ0);  aZ1 = wmma_bf16(a1, bZ, aZ1);
          aI0 = wmma_bf16(a0, bN, aI0);  aI1 = wmma_bf16(a1, bN, aI1);
        }
      }
      { // gh part: K = 512 over h_{t-1} (16 K-tiles)
        const unsigned short* pR = whhP + (((nt)      * 16) << 9) + laneB;
        const unsigned short* pZ = whhP + (((nt + 32) * 16) << 9) + laneB;
        const unsigned short* pN = whhP + (((nt + 64) * 16) << 9) + laneB;
#pragma unroll 1
        for (int kt = 0; kt < 16; ++kt) {
          const int o = kt << 9;
          v16bf a0 = ld16(hT0 + o);
          v16bf a1 = ld16(hT1 + o);
          v16bf bR = ld16(pR + o), bZ = ld16(pZ + o), bN = ld16(pN + o);
          aR0 = wmma_bf16(a0, bR, aR0);  aR1 = wmma_bf16(a1, bR, aR1);
          aZ0 = wmma_bf16(a0, bZ, aZ0);  aZ1 = wmma_bf16(a1, bZ, aZ1);
          aH0 = wmma_bf16(a0, bN, aH0);  aH1 = wmma_bf16(a1, bN, aH1);
        }
      }

#pragma unroll
      for (int m2 = 0; m2 < 2; ++m2) {
        v8f accR = m2 ? aR1 : aR0;
        v8f accZ = m2 ? aZ1 : aZ0;
        v8f accI = m2 ? aI1 : aI0;
        v8f accH = m2 ? aH1 : aH0;
        v8f h = h_reg[m2][ns];
#pragma unroll
        for (int i = 0; i < 8; ++i) {
          float r   = 1.0f / (1.0f + __expf(-accR[i]));
          float z   = 1.0f / (1.0f + __expf(-accZ[i]));
          float pre = accI[i] + r * accH[i];
          float e2  = __expf(-2.0f * pre);
          float nn  = (1.0f - e2) / (1.0f + e2);     // tanh(pre)
          h[i] = (1.0f - z) * nn + z * h[i];
        }
        h_reg[m2][ns] = h;
      }
    }
    __syncthreads();   // everyone done reading hA_s(t-1) / xA_s(t)

    // ---- write h_t (bf16, A-layout) + stage x_{t+1} ----
#pragma unroll
    for (int ns = 0; ns < 4; ++ns) {
      int kG = nb + (ns << 4) + lo16;          // hidden index = K-dim of hA
      int kt = kG >> 5, kk = kG & 31;
      int lhalf = ((kk >> 3) & 1) << 4;
      int j = (kk & 7) | ((kk >> 4) << 3);
#pragma unroll
      for (int m2 = 0; m2 < 2; ++m2) {
#pragma unroll
        for (int i = 0; i < 8; ++i) {
          int mm = i + ((lane >> 4) << 3);     // row within M-tile (C layout)
          hA_s[(((m2 << 4) + kt) << 9) + ((mm + lhalf) << 4) + j] =
              f2bf(h_reg[m2][ns][i]);
        }
      }
    }
    if (t + 1 < SEQ) stage_x(x, xA_s, bBase, t + 1, tid);
    __syncthreads();
  }

  // ---- out = h_49 @ w_out^T + b_out  (32x512 @ 512x256) ----
  {
    const int nbo = wv << 5;                   // unique 32-col output chunk
#pragma unroll
    for (int ns = 0; ns < 2; ++ns) {
      int n0 = nbo + (ns << 4);
      float bo = b_out[n0 + lo16];
      v8f acc0 = splat8(bo), acc1 = splat8(bo);
      const unsigned short* pO = woutP + (((n0 >> 4) * 16) << 9) + laneB;
#pragma unroll 1
      for (int kt = 0; kt < 16; ++kt) {
        const int o = kt << 9;
        v16bf a0 = ld16(hT0 + o);
        v16bf a1 = ld16(hT1 + o);
        v16bf b  = ld16(pO + o);
        acc0 = wmma_bf16(a0, b, acc0);
        acc1 = wmma_bf16(a1, b, acc1);
      }
      int col   = n0 + lo16;
      int rhalf = (lane >> 4) << 3;
#pragma unroll
      for (int m2 = 0; m2 < 2; ++m2) {
        v8f acc = m2 ? acc1 : acc0;
#pragma unroll
        for (int i = 0; i < 8; ++i) {
          int row = bBase + (m2 << 4) + rhalf + i;
          out[row * IN_DIM + col] = acc[i];
        }
      }
    }
  }
}

extern "C" void kernel_launch(void* const* d_in, const int* in_sizes, int n_in,
                              void* d_out, int out_size, void* d_ws, size_t ws_size,
                              hipStream_t stream) {
  const float* x     = (const float*)d_in[0];
  const float* w_ih  = (const float*)d_in[1];
  const float* w_hh  = (const float*)d_in[2];
  const float* b_ih  = (const float*)d_in[3];
  const float* b_hh  = (const float*)d_in[4];
  const float* w_out = (const float*)d_in[5];
  const float* b_out = (const float*)d_in[6];

  // Packed bf16 weights in workspace: 768KB + 1.5MB + 256KB = ~2.5MB.
  unsigned short* wihP  = (unsigned short*)d_ws;
  unsigned short* whhP  = wihP + 1536 * 256;
  unsigned short* woutP = whhP + 1536 * 512;

  pack_w<<<(1536 * 128 + 255) / 256, 256, 0, stream>>>(w_ih,  (unsigned*)wihP,  1536, 256);
  pack_w<<<(1536 * 256 + 255) / 256, 256, 0, stream>>>(w_hh,  (unsigned*)whhP,  1536, 512);
  pack_w<<<(256  * 256 + 255) / 256, 256, 0, stream>>>(w_out, (unsigned*)woutP, 256,  512);

  gru_fused<<<BATCH / TILE_B, 256, 0, stream>>>(x, b_ih, b_hh, b_out,
                                                wihP, whhP, woutP, (float*)d_out);
}
